// SSA_72438918414603
// MI455X (gfx1250) — compile-verified, matrix operand order checked by reading, not run on hardware
//
#include <hip/hip_runtime.h>

typedef __attribute__((ext_vector_type(16))) _Float16 v16h;
typedef __attribute__((ext_vector_type(8)))  _Float16 v8h;
typedef __attribute__((ext_vector_type(8)))  float    v8f;

#define CCH 512
#define HWP 1024
#define CHW (CCH*HWP)
#define NPIX 65536   /* B*H*W = 64*1024 */

// ---------------- kernel 1: weight prep (fold sieve weights, f32 -> f16) ----
__global__ void snn_prep_weights(const float* __restrict__ w1,
                                 const float* __restrict__ w2,
                                 const float* __restrict__ wsrc,
                                 const float* __restrict__ wlin,
                                 _Float16* __restrict__ wc16,
                                 _Float16* __restrict__ ws16,
                                 _Float16* __restrict__ wl16) {
  int i = blockIdx.x * blockDim.x + threadIdx.x;   // 0 .. 512*512-1
  wc16[i] = (_Float16)(w1[i] + w2[i]);   // conv(x,w1)+conv(x,w2) == conv(x,w1+w2)
  ws16[i] = (_Float16)wsrc[i];
  wl16[i] = (_Float16)wlin[i];
}

// ---------------- kernel 2: binarize + transpose to pixel-major f16 --------
// x[b][c][hw] (f32) -> apack[b*1024+hw][c] (f16, {0,1})
__global__ void snn_pack_spikes(const float* __restrict__ x,
                                _Float16* __restrict__ apack) {
  __shared__ _Float16 tile[32][33];           // +1 pad: no bank conflicts
  const int b  = blockIdx.z;                  // 0..63
  const int ct = blockIdx.y;                  // 0..15  (channel tile of 32)
  const int ht = blockIdx.x;                  // 0..31  (hw tile of 32)
  const int t   = threadIdx.x;                // 256 threads
  const int col = t & 31;
  const int rb  = t >> 5;                     // 8 row-groups
#pragma unroll
  for (int i = 0; i < 4; ++i) {
    int row = rb + i * 8;                     // channel-local
    float v = x[(size_t)b * CHW + (size_t)(ct * 32 + row) * HWP + ht * 32 + col];
    tile[row][col] = (v >= 1.0f) ? (_Float16)1.0f : (_Float16)0.0f;
  }
  __syncthreads();
#pragma unroll
  for (int i = 0; i < 4; ++i) {
    int row = rb + i * 8;                     // hw-local
    apack[(size_t)(b * HWP + ht * 32 + row) * CCH + ct * 32 + col] = tile[col][row];
  }
}

// ---------------- kernel 3: fused triple-GEMM SNN block --------------------
// Per WG: 64 pixels. 8 waves: wave = (rbk in 0..3 row-block of 16) x (half of 256 couts).
__global__ __launch_bounds__(256, 1) void snn_gemm(
    const _Float16* __restrict__ apack,
    const _Float16* __restrict__ wc,     // w_sieve1 + w_sieve2 (f16)
    const _Float16* __restrict__ wsrc,   // w_source (f16)
    const _Float16* __restrict__ wlin,   // w_lin (f16)
    float* __restrict__ out) {
  __shared__ _Float16 gated[64 * CCH];   // 64 KB binary gated activations
  __shared__ float otile[8][16][17];     // per-wave 16x16 f32 transpose (+pad)

  const int wg    = blockIdx.x;          // 0..1023
  const int lane  = threadIdx.x & 31;
  const int wave  = threadIdx.x >> 5;    // 0..7
  const int rbk   = wave & 3;            // row block (16 pixels)
  const int half  = wave >> 2;           // cout half (256 channels)
  const int lrow  = lane & 15;
  const int hi    = lane >> 4;           // lane half (0/1)
  const int khalf = hi ? 8 : 0;          // ISA 16-bit A/B fragment K split
  const int mrow  = wg * 64 + rbk * 16 + lrow;   // global A row for this lane
  const int mbase = rbk * 16 + (hi ? 8 : 0);     // C/D row base (M = r + 8*hi)

  // ---------------- phase 1: sieve + source GEMMs, threshold, gate ----------
  for (int nt = 0; nt < 16; ++nt) {
    const int n = half * 256 + nt * 16 + lrow;   // output channel (B column)
    v8f acc_sv = {};
    v8f acc_sc = {};
    for (int kc = 0; kc < 16; ++kc) {
      const int kb = kc * 32 + khalf;
      const v8h a0 = *(const v8h*)(apack + (size_t)mrow * CCH + kb);
      const v8h a1 = *(const v8h*)(apack + (size_t)mrow * CCH + kb + 16);
      const v8h c0 = *(const v8h*)(wc   + (size_t)n * CCH + kb);
      const v8h c1 = *(const v8h*)(wc   + (size_t)n * CCH + kb + 16);
      const v8h s0 = *(const v8h*)(wsrc + (size_t)n * CCH + kb);
      const v8h s1 = *(const v8h*)(wsrc + (size_t)n * CCH + kb + 16);
      v16h a, bsv, bsc;
#pragma unroll
      for (int i = 0; i < 8; ++i) {
        a[i] = a0[i];   a[i + 8] = a1[i];
        bsv[i] = c0[i]; bsv[i + 8] = c1[i];
        bsc[i] = s0[i]; bsc[i + 8] = s1[i];
      }
      acc_sv = __builtin_amdgcn_wmma_f32_16x16x32_f16(false, a, false, bsv,
                                                      (short)0, acc_sv, false, false);
      acc_sc = __builtin_amdgcn_wmma_f32_16x16x32_f16(false, a, false, bsc,
                                                      (short)0, acc_sc, false, false);
    }
    // gated = (sieve_v >= 4) & (source_v >= 1) — write binary f16 to LDS
#pragma unroll
    for (int r = 0; r < 8; ++r) {
      _Float16 g = (acc_sv[r] >= 4.0f && acc_sc[r] >= 1.0f)
                       ? (_Float16)1.0f : (_Float16)0.0f;
      gated[(mbase + r) * CCH + n] = g;
    }
  }

  __syncthreads();

  // ---------------- phase 2: out = 0.5*(gated @ w_lin^T + x_in) -------------
  const int mgbase = wg * 64 + rbk * 16;       // 16-aligned pixel base of tile
  const int bidx   = mgbase >> 10;             // batch (constant across tile)
  const int hwbase = mgbase & 1023;            // hw base (16-aligned)

  for (int nt = 0; nt < 16; ++nt) {
    const int n = half * 256 + nt * 16 + lrow;
    v8f acc = {};
    for (int kc = 0; kc < 16; ++kc) {
      const int kb = kc * 32 + khalf;
      const v8h a0 = *(const v8h*)(&gated[(rbk * 16 + lrow) * CCH + kb]);
      const v8h a1 = *(const v8h*)(&gated[(rbk * 16 + lrow) * CCH + kb + 16]);
      const v8h b0 = *(const v8h*)(wlin + (size_t)n * CCH + kb);
      const v8h b1 = *(const v8h*)(wlin + (size_t)n * CCH + kb + 16);
      v16h a, bb;
#pragma unroll
      for (int i = 0; i < 8; ++i) {
        a[i] = a0[i];  a[i + 8] = a1[i];
        bb[i] = b0[i]; bb[i + 8] = b1[i];
      }
      acc = __builtin_amdgcn_wmma_f32_16x16x32_f16(false, a, false, bb,
                                                   (short)0, acc, false, false);
    }
    // residual add in native fragment layout (coalesced apack f16 read),
    // then transpose the 16x16 f32 tile through LDS for coalesced stores.
#pragma unroll
    for (int r = 0; r < 8; ++r) {
      const int mloc = mbase + r;                       // WG-local row 0..63
      const int mg   = wg * 64 + mloc;
      const float resid = (float)apack[(size_t)mg * CCH + n];  // binary x_in
      otile[wave][mloc - rbk * 16][lrow] = 0.5f * (acc[r] + resid);
    }
    __builtin_amdgcn_s_barrier_signal(-1);   // ds-wait + workgroup sync point
    __builtin_amdgcn_s_barrier_wait(-1);
    // transposed read-out: lanes 0..15 walk hw for channel group 0..7,
    // lanes 16..31 walk hw for channel group 8..15 -> full 64B store lines
    const int cbase = hi * 8;
    const int nb = half * 256 + nt * 16;
#pragma unroll
    for (int j = 0; j < 8; ++j) {
      const int c = nb + cbase + j;
      out[(size_t)bidx * CHW + (size_t)c * HWP + hwbase + lrow] =
          otile[wave][lrow][cbase + j];
    }
  }
}

extern "C" void kernel_launch(void* const* d_in, const int* in_sizes, int n_in,
                              void* d_out, int out_size, void* d_ws, size_t ws_size,
                              hipStream_t stream) {
  (void)in_sizes; (void)n_in; (void)out_size; (void)ws_size;
  const float* x   = (const float*)d_in[0];
  const float* w1  = (const float*)d_in[1];
  const float* w2  = (const float*)d_in[2];
  const float* wso = (const float*)d_in[3];
  const float* wl  = (const float*)d_in[4];
  float* out = (float*)d_out;

  // workspace layout: [A_pack 64MB f16][w_comb 512KB][w_src 512KB][w_lin 512KB]
  char* ws = (char*)d_ws;
  _Float16* apack = (_Float16*)ws;
  _Float16* wc16  = (_Float16*)(ws + (size_t)NPIX * CCH * sizeof(_Float16));
  _Float16* ws16  = wc16 + (size_t)CCH * CCH;
  _Float16* wl16  = ws16 + (size_t)CCH * CCH;

  snn_prep_weights<<<(CCH * CCH) / 256, 256, 0, stream>>>(w1, w2, wso, wl,
                                                          wc16, ws16, wl16);
  dim3 pgrid(32, 16, 64);   // hw-tiles, c-tiles, batch
  snn_pack_spikes<<<pgrid, 256, 0, stream>>>(x, apack);
  snn_gemm<<<NPIX / 64, 256, 0, stream>>>(apack, wc16, ws16, wl16, out);
}
